// AdditiveAttention_41626823032957
// MI455X (gfx1250) — matrix operand order repaired and checked by reading, not compiled, not run on hardware
//
#include <hip/hip_runtime.h>
#include <hip/hip_bf16.h>

#define Bb_   4
#define LQ_   256
#define LK_   512
#define DQ_   256
#define DK_   256
#define DV_   256
#define H_    128

typedef __attribute__((ext_vector_type(16))) _Float16 v16h;
typedef __attribute__((ext_vector_type(8)))  _Float16 v8h;
typedef __attribute__((ext_vector_type(8)))  float    v8f;

// ---------------------------------------------------------------------------
// fp32 -> fp16 elementwise convert
// ---------------------------------------------------------------------------
__global__ void cvt_f32_f16_kernel(const float* __restrict__ in,
                                   _Float16* __restrict__ out, int n) {
    int i = blockIdx.x * blockDim.x + threadIdx.x;
    if (i < n) out[i] = (_Float16)in[i];
}

// ---------------------------------------------------------------------------
// Tiled transpose + fp32->fp16 convert: in[R,C] f32 -> out[C,R] f16.
// 32x32 LDS tile, block (32,8), batch via blockIdx.z.
// ---------------------------------------------------------------------------
__global__ void transpose_cvt_kernel(const float* __restrict__ in,
                                     _Float16* __restrict__ out,
                                     int R, int C, long strideIn, long strideOut) {
    __shared__ _Float16 tile[32][33];   // +1 pad to dodge bank conflicts
    const float* inb  = in  + (long)blockIdx.z * strideIn;
    _Float16*    outb = out + (long)blockIdx.z * strideOut;
    const int c0 = blockIdx.x * 32;
    const int r0 = blockIdx.y * 32;
    const int tx = threadIdx.x, ty = threadIdx.y;
#pragma unroll
    for (int i = 0; i < 32; i += 8)
        tile[ty + i][tx] = (_Float16)inb[(long)(r0 + ty + i) * C + (c0 + tx)];
    __syncthreads();
#pragma unroll
    for (int i = 0; i < 32; i += 8)
        outb[(long)(c0 + ty + i) * R + (r0 + tx)] = tile[tx][ty + i];
}

// ---------------------------------------------------------------------------
// f16 WMMA GEMM, NT form: C[M,N] (f32) = A[M,K] (f16 row-major) * BT[N,K]^T.
// One wave32 per 16x64 output strip (4 n-tiles sharing the A fragment),
// 4 waves per block. Batch via blockIdx.y (element strides).
// Fragment layouts used here (contiguous k, same for A and B; wave32), lane L: g=L/16, r=L%16:
//   A frag half j : m = r,      k = j + 16g                -> two 16B loads
//   B frag half j : n = r,      k = j + 16g                -> one 32B load/tile
//   C/D  vgpr rr  : m = rr+8g,  n = r
// ---------------------------------------------------------------------------
__global__ void gemm_f16_wmma_nt_kernel(const _Float16* __restrict__ A,
                                        const _Float16* __restrict__ BT,
                                        float* __restrict__ C,
                                        int M, int N, int K,
                                        long sA, long sB, long sC) {
    const int wave = threadIdx.x >> 5;
    const int lane = threadIdx.x & 31;
    const int nStrips = N >> 6;                     // strips of 64 columns
    const int strip   = blockIdx.x * 4 + wave;
    if (strip >= (M >> 4) * nStrips) return;

    const int tm = (strip / nStrips) << 4;
    const int tn = (strip % nStrips) << 6;
    const int g  = lane >> 4;
    const int r  = lane & 15;

    const _Float16* Ab = A  + (long)blockIdx.y * sA;
    const _Float16* Bb = BT + (long)blockIdx.y * sB;
    float*          Cb = C  + (long)blockIdx.y * sC;

    const _Float16* arow = Ab + (long)(tm + r) * K;         // A row for this lane
    const _Float16* brow = Bb + (long)(tn + r) * K;         // BT row, n-tile 0
    const long tstride = (long)16 * K;                      // next n-tile in BT

    v8f acc0 = {}, acc1 = {}, acc2 = {}, acc3 = {};

    for (int k0 = 0; k0 < K; k0 += 32) {
        // ---- A fragment: two contiguous 16-byte loads ----
        const _Float16* ap = arow + k0 + 16 * g;
        v8h a_lo = *(const v8h*)(ap);         // k = k0+16g+0..7
        v8h a_hi = *(const v8h*)(ap + 8);     // k = k0+16g+8..15
        v16h af;
#pragma unroll
        for (int j = 0; j < 8; ++j) { af[j] = a_lo[j]; af[j + 8] = a_hi[j]; }

        // ---- B fragments: one contiguous 32-byte load per n-tile ----
        const _Float16* bp = brow + k0 + 16 * g;
        v16h b0 = *(const v16h*)(bp);
        v16h b1 = *(const v16h*)(bp + tstride);
        v16h b2 = *(const v16h*)(bp + 2 * tstride);
        v16h b3 = *(const v16h*)(bp + 3 * tstride);

        acc0 = __builtin_amdgcn_wmma_f32_16x16x32_f16(false, af, false, b0,
                                                      (short)0, acc0, false, false);
        acc1 = __builtin_amdgcn_wmma_f32_16x16x32_f16(false, af, false, b1,
                                                      (short)0, acc1, false, false);
        acc2 = __builtin_amdgcn_wmma_f32_16x16x32_f16(false, af, false, b2,
                                                      (short)0, acc2, false, false);
        acc3 = __builtin_amdgcn_wmma_f32_16x16x32_f16(false, af, false, b3,
                                                      (short)0, acc3, false, false);
    }

#pragma unroll
    for (int rr = 0; rr < 8; ++rr) {
        float* crow = Cb + (long)(tm + rr + 8 * g) * N + tn + r;
        crow[0]  = acc0[rr];
        crow[16] = acc1[rr];
        crow[32] = acc2[rr];
        crow[48] = acc3[rr];
    }
}

// ---------------------------------------------------------------------------
// Additive-attention scores: scores[b,q,lk] = sum_h wv[h]*tanh(qp[b,q,h]+kp[b,lk,h])
// One workgroup (256 thr = 8 waves) per (b, q). Wave w handles lk = w, w+8, ...
// Key rows are streamed global->LDS with global_load_async_to_lds_b128
// (wave-private double buffer, ASYNCcnt-synchronized) so the L2 latency hides
// behind the ~128 TRANS ops per row. Each lane moves 16B (32 lanes = 512B row),
// then reduces 4 h-values; cross-lane sum via shfl_xor (wave32).
// This is the roofline-limiting stage (~67M v_tanh_f32 TRANS ops).
// ---------------------------------------------------------------------------
__global__ void scores_kernel(const float* __restrict__ qp,
                              const float* __restrict__ kp,
                              const float* __restrict__ wv,
                              float* __restrict__ scores) {
    const int bq = blockIdx.x;            // [0, B*LQ)
    const int b  = bq / LQ_;

    __shared__ float qs[H_];
    __shared__ float ws[H_];
    __shared__ float kbuf[8][2][H_];      // per-wave double buffer (8KB)

    for (int i = threadIdx.x; i < H_; i += blockDim.x) {
        qs[i] = qp[(long)bq * H_ + i];
        ws[i] = wv[i];
    }
    __syncthreads();

    const int wave = threadIdx.x >> 5;
    const int lane = threadIdx.x & 31;
    const float* kb = kp + (long)b * LK_ * H_;

    // Generic-pointer low 32 bits of a __shared__ address == LDS byte offset
    // (ISA 10.2 aperture rules) -- exactly what the async op's VDST wants.
    const unsigned lds0 = (unsigned)(size_t)(&kbuf[wave][0][4 * lane]);
    const unsigned lds1 = (unsigned)(size_t)(&kbuf[wave][1][4 * lane]);

    // prime buffer 0 with row lk = wave
    {
        const float* g = kb + (long)wave * H_ + 4 * lane;
        asm volatile("global_load_async_to_lds_b128 %0, %1, off"
                     :: "v"(lds0), "v"(g) : "memory");
    }

    int buf = 0;
    for (int lk = wave; lk < LK_; lk += 8) {
        if (lk + 8 < LK_) {
            const float* gn  = kb + (long)(lk + 8) * H_ + 4 * lane;
            unsigned ldsn = buf ? lds0 : lds1;
            asm volatile("global_load_async_to_lds_b128 %0, %1, off"
                         :: "v"(ldsn), "v"(gn) : "memory");
            // async loads complete in order: cnt<=1 => older buffer resident
            asm volatile("s_wait_asynccnt 0x1" ::: "memory");
        } else {
            asm volatile("s_wait_asynccnt 0x0" ::: "memory");
        }

        const float* krow = kbuf[wave][buf];
        float s = 0.f;
#pragma unroll
        for (int h0 = 0; h0 < H_; h0 += 32) {
            int h = h0 + lane;
            s += ws[h] * tanhf(qs[h] + krow[h]);        // v_tanh_f32 (TRANS32)
        }
#pragma unroll
        for (int off = 16; off > 0; off >>= 1)
            s += __shfl_xor(s, off, 32);
        if (lane == 0) scores[(long)bq * LK_ + lk] = s;
        buf ^= 1;
    }
}

// ---------------------------------------------------------------------------
// Masked softmax over LK, emit fp16 attention weights.
// One wave32 per row; 16 elements per lane; shuffle max/sum reductions.
// ---------------------------------------------------------------------------
__global__ void softmax_kernel(const float* __restrict__ scores,
                               const int* __restrict__ valid_lens,
                               _Float16* __restrict__ attn) {
    const int row  = blockIdx.x * 8 + (threadIdx.x >> 5);   // [0, B*LQ)
    const int lane = threadIdx.x & 31;
    const int b    = row / LQ_;
    const int vl   = valid_lens[b];
    const float* srow = scores + (long)row * LK_;

    float vals[LK_ / 32];
    float m = -1e30f;
#pragma unroll
    for (int i = 0; i < LK_ / 32; ++i) {
        int lk = lane + 32 * i;
        float s = (lk < vl) ? srow[lk] : -1e6f;
        vals[i] = s;
        m = fmaxf(m, s);
    }
#pragma unroll
    for (int off = 16; off > 0; off >>= 1)
        m = fmaxf(m, __shfl_xor(m, off, 32));

    float sum = 0.f;
#pragma unroll
    for (int i = 0; i < LK_ / 32; ++i) {
        vals[i] = __expf(vals[i] - m);
        sum += vals[i];
    }
#pragma unroll
    for (int off = 16; off > 0; off >>= 1)
        sum += __shfl_xor(sum, off, 32);

    const float inv = 1.f / sum;
    _Float16* arow = attn + (long)row * LK_;
#pragma unroll
    for (int i = 0; i < LK_ / 32; ++i)
        arow[lane + 32 * i] = (_Float16)(vals[i] * inv);
}

// ---------------------------------------------------------------------------
extern "C" void kernel_launch(void* const* d_in, const int* in_sizes, int n_in,
                              void* d_out, int out_size, void* d_ws, size_t ws_size,
                              hipStream_t stream) {
    const float* queries = (const float*)d_in[0];  // [B,LQ,DQ]
    const float* keys    = (const float*)d_in[1];  // [B,LK,DK]
    const float* values  = (const float*)d_in[2];  // [B,LK,DV]
    const float* Wq      = (const float*)d_in[3];  // [DQ,H]
    const float* Wk      = (const float*)d_in[4];  // [DK,H]
    const float* wv      = (const float*)d_in[5];  // [H]
    const int*   vlens   = (const int*)d_in[6];    // [B]
    float* out = (float*)d_out;                    // [B,LQ,DV]

    // ---- workspace carving (256B aligned) ----
    char* p = (char*)d_ws;
    auto carve = [&](size_t bytes) -> char* {
        char* r = p;
        p += (bytes + 255) & ~(size_t)255;
        return r;
    };
    _Float16* qf16 = (_Float16*)carve((size_t)Bb_ * LQ_ * DQ_ * 2);  // [B*LQ, DQ]
    _Float16* kf16 = (_Float16*)carve((size_t)Bb_ * LK_ * DK_ * 2);  // [B*LK, DK]
    _Float16* WqT  = (_Float16*)carve((size_t)H_ * DQ_ * 2);         // [H, DQ]
    _Float16* WkT  = (_Float16*)carve((size_t)H_ * DK_ * 2);         // [H, DK]
    _Float16* vT   = (_Float16*)carve((size_t)Bb_ * DV_ * LK_ * 2);  // [B][DV, LK]
    float*    q32  = (float*)carve((size_t)Bb_ * LQ_ * H_ * 4);
    float*    k32  = (float*)carve((size_t)Bb_ * LK_ * H_ * 4);
    float*    sc32 = (float*)carve((size_t)Bb_ * LQ_ * LK_ * 4);
    _Float16* attn = (_Float16*)carve((size_t)Bb_ * LQ_ * LK_ * 2);  // [B*LQ, LK]

    // ---- stage 0a: fp32 -> fp16 converts (A matrices) ----
    {
        int n = Bb_ * LQ_ * DQ_;
        cvt_f32_f16_kernel<<<(n + 255) / 256, 256, 0, stream>>>(queries, qf16, n);
        n = Bb_ * LK_ * DK_;
        cvt_f32_f16_kernel<<<(n + 255) / 256, 256, 0, stream>>>(keys, kf16, n);
    }
    // ---- stage 0b: transpose+convert (B matrices -> BT f16) ----
    transpose_cvt_kernel<<<dim3(H_ / 32, DQ_ / 32, 1), dim3(32, 8), 0, stream>>>(
        Wq, WqT, DQ_, H_, 0, 0);
    transpose_cvt_kernel<<<dim3(H_ / 32, DK_ / 32, 1), dim3(32, 8), 0, stream>>>(
        Wk, WkT, DK_, H_, 0, 0);
    transpose_cvt_kernel<<<dim3(DV_ / 32, LK_ / 32, Bb_), dim3(32, 8), 0, stream>>>(
        values, vT, LK_, DV_, (long)LK_ * DV_, (long)DV_ * LK_);

    // ---- stage 1: projections via WMMA (NT) ----
    {   // q32[B*LQ, H] = qf16[B*LQ, DQ] * WqT[H, DQ]^T
        int M = Bb_ * LQ_, N = H_, K = DQ_;
        int strips = (M / 16) * (N / 64);
        gemm_f16_wmma_nt_kernel<<<dim3(strips / 4, 1), 128, 0, stream>>>(
            qf16, WqT, q32, M, N, K, 0, 0, 0);
    }
    {   // k32[B*LK, H] = kf16[B*LK, DK] * WkT[H, DK]^T
        int M = Bb_ * LK_, N = H_, K = DK_;
        int strips = (M / 16) * (N / 64);
        gemm_f16_wmma_nt_kernel<<<dim3(strips / 4, 1), 128, 0, stream>>>(
            kf16, WkT, k32, M, N, K, 0, 0, 0);
    }

    // ---- stage 2: additive scores (tanh, TRANS-bound, async-LDS streamed) ----
    scores_kernel<<<Bb_ * LQ_, 256, 0, stream>>>(q32, k32, wv, sc32);

    // ---- stage 3: masked softmax -> f16 attention ----
    softmax_kernel<<<(Bb_ * LQ_) / 8, 256, 0, stream>>>(sc32, vlens, attn);

    // ---- stage 4: out[b] = attn[b] * values[b] via batched WMMA (NT) ----
    {
        int M = LQ_, N = DV_, K = LK_;
        int strips = (M / 16) * (N / 64);
        gemm_f16_wmma_nt_kernel<<<dim3(strips / 4, Bb_), 128, 0, stream>>>(
            attn, vT, out, M, N, K,
            (long)LQ_ * LK_, (long)DV_ * LK_, (long)LQ_ * DV_);
    }
}
